// LKAAttention_49709951484567
// MI455X (gfx1250) — matrix operand order repaired and checked by reading, not compiled, not run on hardware
//
#include <hip/hip_runtime.h>
#include <hip/hip_bf16.h>

typedef __attribute__((ext_vector_type(16))) __bf16 v16bf;
typedef __attribute__((ext_vector_type(8)))  __bf16 v8bf;
typedef __attribute__((ext_vector_type(4)))  __bf16 v4bf;
typedef __attribute__((ext_vector_type(8)))  float  v8f;

#define B_   16
#define S_   4096
#define HID_ 768
#define H_   12
#define DH_  64

__device__ __forceinline__ v8f zero8() {
  v8f z;
#pragma unroll
  for (int i = 0; i < 8; ++i) z[i] = 0.0f;
  return z;
}

__device__ __forceinline__ v8f wmma_bf16(v16bf a, v16bf b, v8f c) {
  // D = A(16x32 bf16) * B(32x16 bf16) + C(16x16 f32)
  return __builtin_amdgcn_wmma_f32_16x16x32_bf16(false, a, false, b, (short)0, c,
                                                 false, false);
}

// Build a 16-element bf16 fragment from two 16-byte aligned LDS chunks.
__device__ __forceinline__ v16bf frag16(const __bf16* p0, const __bf16* p1) {
  v8bf lo = *(const v8bf*)p0;
  v8bf hi = *(const v8bf*)p1;
  v16bf r;
#pragma unroll
  for (int e = 0; e < 8; ++e) { r[e] = lo[e]; r[8 + e] = hi[e]; }
  return r;
}

// LDS-relative byte offset of a shared-memory pointer: generic (flat) shared
// addresses are {aperture_hi, lds_offset} so the low 32 bits are the offset.
__device__ __forceinline__ unsigned ldsOff(const void* p) {
  return (unsigned)(uintptr_t)p;
}

// CDNA5 async copy global -> LDS (GVS mode), tracked by ASYNCcnt.
__device__ __forceinline__ void asyncLdsB128(unsigned lds, const void* gbase,
                                             unsigned voff) {
  asm volatile("global_load_async_to_lds_b128 %0, %1, %2"
               :
               : "v"(lds), "v"(voff), "s"((unsigned long long)(uintptr_t)gbase)
               : "memory");
}
__device__ __forceinline__ void waitAsync0() {
  asm volatile("s_wait_asynccnt 0" ::: "memory");
}

// ---------------------------------------------------------------------------
// Kernel 1: fused QKV projection. grid = (12 Ntiles, 512 Mtiles, 3 {q,k,v})
// C[m,n] = (x @ W + b)[m,n] * mask[m]; q,k get elu+1 and 1/sqrt(D) scaling.
// Q -> [B,H,S,D], K -> [B,H,D,S] (transposed for kernel 2), V -> [B,H,S,D]
// ---------------------------------------------------------------------------
__global__ __launch_bounds__(256) void k_qkv(
    const float* __restrict__ x, const float* __restrict__ mask,
    const float* __restrict__ Wq, const float* __restrict__ bq,
    const float* __restrict__ Wk, const float* __restrict__ bk,
    const float* __restrict__ Wv, const float* __restrict__ bv,
    __bf16* __restrict__ Qb, __bf16* __restrict__ Kt, __bf16* __restrict__ Vb) {
  const int z  = blockIdx.z;
  const float* W    = (z == 0) ? Wq : (z == 1) ? Wk : Wv;
  const float* bias = (z == 0) ? bq : (z == 1) ? bk : bv;
  const int N0 = blockIdx.x * 64;
  const int M0 = blockIdx.y * 128;
  const int t = threadIdx.x, lane = t & 31, w = t >> 5;

  __shared__ __bf16 As[128 * 40];  // A tile, row-major [m][k], pad 40
  __shared__ __bf16 Bs[64 * 40];   // B tile, transposed [n][k], pad 40

  v8f acc[4];
#pragma unroll
  for (int i = 0; i < 4; ++i) acc[i] = zero8();

  const int h8 = (lane >> 4) * 8;   // unused; A and B fragments both use kb
  const int kb = (lane >> 4) * 16;  // B fragment K sub-base

  for (int k0 = 0; k0 < HID_; k0 += 32) {
    // prefetch next weight tile (warms WGP$/L2 ahead of the converting stage)
    if (k0 + 32 < HID_) {
      int pr = t >> 3, pc = (t & 7) * 8;
      __builtin_prefetch(W + (size_t)(k0 + 32 + pr) * HID_ + N0 + pc, 0, 3);
    }
    __syncthreads();
    // Stage A: 128x32 fp32 -> bf16 (1024 float4 slots, 4 per thread)
#pragma unroll
    for (int i = 0; i < 4; ++i) {
      int slot = t + 256 * i;
      int row = slot >> 3, c4 = slot & 7;
      float4 v = *(const float4*)(x + (size_t)(M0 + row) * HID_ + k0 + c4 * 4);
      __bf16* d = &As[row * 40 + c4 * 4];
      d[0] = (__bf16)v.x; d[1] = (__bf16)v.y; d[2] = (__bf16)v.z; d[3] = (__bf16)v.w;
    }
    // Stage B transposed: 32x64 fp32 -> Bs[n][k]
#pragma unroll
    for (int i = 0; i < 2; ++i) {
      int slot = t + 256 * i;
      int kk = slot >> 4, n4 = slot & 15;
      float4 v = *(const float4*)(W + (size_t)(k0 + kk) * HID_ + N0 + n4 * 4);
      Bs[(n4 * 4 + 0) * 40 + kk] = (__bf16)v.x;
      Bs[(n4 * 4 + 1) * 40 + kk] = (__bf16)v.y;
      Bs[(n4 * 4 + 2) * 40 + kk] = (__bf16)v.z;
      Bs[(n4 * 4 + 3) * 40 + kk] = (__bf16)v.w;
    }
    __syncthreads();
    const __bf16* arow = &As[(w * 16 + (lane & 15)) * 40];
    v16bf a = frag16(arow + kb, arow + kb + 8);
    v16bf bfr[4];
#pragma unroll
    for (int nt = 0; nt < 4; ++nt) {
      const __bf16* bcol = &Bs[(nt * 16 + (lane & 15)) * 40];
      bfr[nt] = frag16(bcol + kb, bcol + kb + 8);
    }
#pragma unroll
    for (int nt = 0; nt < 4; ++nt) acc[nt] = wmma_bf16(a, bfr[nt], acc[nt]);
  }

  const int h = N0 >> 6;  // head index (N0 multiple of 64)
  float mrow[8];
#pragma unroll
  for (int j = 0; j < 8; ++j)
    mrow[j] = mask[M0 + w * 16 + j + ((lane >> 4) << 3)];

#pragma unroll
  for (int nt = 0; nt < 4; ++nt) {
    int n = nt * 16 + (lane & 15);  // d within head, 0..63
    float bv_ = bias[N0 + n];
#pragma unroll
    for (int j = 0; j < 8; ++j) {
      int m  = w * 16 + j + ((lane >> 4) << 3);
      int mg = M0 + m;
      int b  = mg >> 12, s = mg & 4095;
      float val = (acc[nt][j] + bv_) * mrow[j];
      if (z < 2) {  // elu(x)+1 then 1/sqrt(64)
        val = (val > 0.0f) ? (val + 1.0f) : __expf(val);
        val *= 0.125f;
      }
      __bf16 o = (__bf16)val;
      size_t bh = (size_t)b * H_ + h;
      if (z == 0)      Qb[(bh * S_ + s) * DH_ + n] = o;
      else if (z == 1) Kt[(bh * DH_ + n) * S_ + s] = o;
      else             Vb[(bh * S_ + s) * DH_ + n] = o;
    }
  }
}

// ---------------------------------------------------------------------------
// Kernel 2: per (b,h): kv = K^T V  (64x64 = A[64x4096] * B[4096x64]) and
// ksum = K^T @ ones, both on the matrix pipe. grid = 192 blocks, 512 threads
// (16 waves; each owns one 16x16 kv tile; the et==0 wave column also
// accumulates ksum against a constant all-ones B fragment -> zero extra
// bytes moved, no separate reduction pass over K).
// ---------------------------------------------------------------------------
__global__ __launch_bounds__(512) void k_kv(
    const __bf16* __restrict__ Kt, const __bf16* __restrict__ Vb,
    __bf16* __restrict__ Kv, __bf16* __restrict__ Ksum) {
  const int bh = blockIdx.x;
  const int t = threadIdx.x, lane = t & 31, w = t >> 5;
  const __bf16* Ktb = Kt + (size_t)bh * DH_ * S_;
  const __bf16* Vbb = Vb + (size_t)bh * S_ * DH_;

  __shared__ __bf16 As[64 * 40];
  __shared__ __bf16 Bs[64 * 40];

  v8f acc = zero8();
  v8f accs = zero8();
  v16bf ones;
#pragma unroll
  for (int e = 0; e < 16; ++e) ones[e] = (__bf16)1.0f;

  const int dt = w >> 2, et = w & 3;
  const int h8 = (lane >> 4) * 8, kb = (lane >> 4) * 16;

  for (int s0 = 0; s0 < S_; s0 += 32) {
    __syncthreads();
    // async stage K^T tile 64(d) x 32(s): 256 x 16B chunks, threads 0..255
    if (t < 256) {
      int row = t >> 2, q = t & 3;  // 4 x 16B per 64B row
      asyncLdsB128(ldsOff(&As[0]) + row * 80 + q * 16, Ktb,
                   (unsigned)(row * (S_ * 2) + s0 * 2 + q * 16));
    }
    {  // stage V tile transposed -> Bs[e][s]
      int ss = t >> 4, e4 = t & 15;
      v4bf v = *(const v4bf*)(Vbb + (size_t)(s0 + ss) * DH_ + e4 * 4);
#pragma unroll
      for (int e = 0; e < 4; ++e) Bs[(e4 * 4 + e) * 40 + ss] = v[e];
    }
    waitAsync0();
    __syncthreads();
    const __bf16* arow = &As[(dt * 16 + (lane & 15)) * 40];
    const __bf16* bcol = &Bs[(et * 16 + (lane & 15)) * 40];
    v16bf a = frag16(arow + kb, arow + kb + 8);
    v16bf b = frag16(bcol + kb, bcol + kb + 8);
    acc = wmma_bf16(a, b, acc);
    if (et == 0) accs = wmma_bf16(a, ones, accs);  // wave-uniform branch
  }
#pragma unroll
  for (int j = 0; j < 8; ++j) {
    int d = dt * 16 + j + ((lane >> 4) << 3);
    int e = et * 16 + (lane & 15);
    Kv[(size_t)bh * DH_ * DH_ + d * DH_ + e] = (__bf16)acc[j];
  }
  // ksum[d] replicated across the 16 accumulator columns; column 0 stores it.
  if (et == 0 && (lane & 15) == 0) {
#pragma unroll
    for (int j = 0; j < 8; ++j) {
      int d = dt * 16 + j + ((lane >> 4) << 3);
      Ksum[bh * DH_ + d] = (__bf16)accs[j];
    }
  }
}

// ---------------------------------------------------------------------------
// Kernel 3: numerator = Q @ kv, denominator via WMMA against a broadcast
// ksum B-tile (lands element-aligned with numerator fragments), then divide.
// Q tile staged with async global->LDS copies.
// grid = (32 s-tiles of 128, 192 bh), 256 threads.
// ---------------------------------------------------------------------------
__global__ __launch_bounds__(256) void k_attn(
    const __bf16* __restrict__ Qb, const __bf16* __restrict__ Kv,
    const __bf16* __restrict__ Ksum, __bf16* __restrict__ Ctx) {
  const int bh = blockIdx.y;
  const int s0 = blockIdx.x * 128;
  const int b = bh / H_, h = bh % H_;
  const int t = threadIdx.x, lane = t & 31, w = t >> 5;
  const __bf16* Qbb = Qb + (size_t)bh * S_ * DH_;

  __shared__ __bf16 As[128 * 72];  // Q tile [m][k], K=64, pad 72
  __shared__ __bf16 Bs[64 * 72];   // kv transposed [e][d]
  __shared__ __bf16 Bd[16 * 72];   // Bd[n][k] = ksum[k] (broadcast columns)

  // async stage Q tile 128x64 bf16: 1024 x 16B chunks, 4 per thread
#pragma unroll
  for (int i = 0; i < 4; ++i) {
    int slot = t + 256 * i;
    int row = slot >> 3, q = slot & 7;  // 8 x 16B per 128B row
    asyncLdsB128(ldsOff(&As[0]) + row * 144 + q * 16, Qbb,
                 (unsigned)((s0 + row) * (DH_ * 2) + q * 16));
  }
#pragma unroll
  for (int i = 0; i < 4; ++i) {  // stage kv -> Bs[e][d]
    int slot = t + 256 * i;
    int d = slot >> 4, e4 = slot & 15;
    v4bf v = *(const v4bf*)(Kv + (size_t)bh * DH_ * DH_ + d * DH_ + e4 * 4);
#pragma unroll
    for (int e = 0; e < 4; ++e) Bs[(e4 * 4 + e) * 72 + d] = v[e];
  }
  if (t < 64) {
    __bf16 v = Ksum[bh * DH_ + t];
#pragma unroll
    for (int n = 0; n < 16; ++n) Bd[n * 72 + t] = v;
  }
  waitAsync0();
  __syncthreads();

  v8f acc[4]; v8f accd = zero8();
#pragma unroll
  for (int i = 0; i < 4; ++i) acc[i] = zero8();
  const int h8 = (lane >> 4) * 8, kbase = (lane >> 4) * 16;
  const __bf16* arow = &As[(w * 16 + (lane & 15)) * 72];
#pragma unroll
  for (int ks = 0; ks < DH_; ks += 32) {
    v16bf a = frag16(arow + ks + kbase, arow + ks + kbase + 8);
    v16bf bfr[4], bd;
#pragma unroll
    for (int nt = 0; nt < 4; ++nt) {
      const __bf16* bcol = &Bs[(nt * 16 + (lane & 15)) * 72 + ks];
      bfr[nt] = frag16(bcol + kbase, bcol + kbase + 8);
    }
    {
      const __bf16* dcol = &Bd[(lane & 15) * 72 + ks];
      bd = frag16(dcol + kbase, dcol + kbase + 8);
    }
#pragma unroll
    for (int nt = 0; nt < 4; ++nt) acc[nt] = wmma_bf16(a, bfr[nt], acc[nt]);
    accd = wmma_bf16(a, bd, accd);
  }

#pragma unroll
  for (int nt = 0; nt < 4; ++nt) {
#pragma unroll
    for (int j = 0; j < 8; ++j) {
      int m = w * 16 + j + ((lane >> 4) << 3);
      int s = s0 + m;
      int n = nt * 16 + (lane & 15);
      float o = acc[nt][j] / accd[j];
      Ctx[((size_t)(b * S_ + s)) * HID_ + h * DH_ + n] = (__bf16)o;
    }
  }
}

// ---------------------------------------------------------------------------
// Kernel 4: out = ctx @ Wo + bo (fp32 output). grid = (12, 512), 256 threads.
// ctx tile (already bf16) staged with async global->LDS copies.
// ---------------------------------------------------------------------------
__global__ __launch_bounds__(256) void k_out(
    const __bf16* __restrict__ Ctx, const float* __restrict__ Wo,
    const float* __restrict__ bo, float* __restrict__ Out) {
  const int N0 = blockIdx.x * 64;
  const int M0 = blockIdx.y * 128;
  const int t = threadIdx.x, lane = t & 31, w = t >> 5;
  __shared__ __bf16 As[128 * 40];
  __shared__ __bf16 Bs[64 * 40];
  v8f acc[4];
#pragma unroll
  for (int i = 0; i < 4; ++i) acc[i] = zero8();
  const int h8 = (lane >> 4) * 8, kb = (lane >> 4) * 16;

  for (int k0 = 0; k0 < HID_; k0 += 32) {
    if (k0 + 32 < HID_) {
      int pr = t >> 3, pc = (t & 7) * 8;
      __builtin_prefetch(Wo + (size_t)(k0 + 32 + pr) * HID_ + N0 + pc, 0, 3);
    }
    __syncthreads();
    // async stage ctx tile 128x32 bf16: 512 x 16B chunks, 2 per thread
#pragma unroll
    for (int i = 0; i < 2; ++i) {
      int slot = t + 256 * i;
      int row = slot >> 2, q = slot & 3;  // 4 x 16B per 64B row
      asyncLdsB128(ldsOff(&As[0]) + row * 80 + q * 16, Ctx,
                   (unsigned)((M0 + row) * (HID_ * 2) + k0 * 2 + q * 16));
    }
#pragma unroll
    for (int i = 0; i < 2; ++i) {  // stage Wo fp32 -> Bs[n][k]
      int slot = t + 256 * i;
      int kk = slot >> 4, n4 = slot & 15;
      float4 v = *(const float4*)(Wo + (size_t)(k0 + kk) * HID_ + N0 + n4 * 4);
      Bs[(n4 * 4 + 0) * 40 + kk] = (__bf16)v.x;
      Bs[(n4 * 4 + 1) * 40 + kk] = (__bf16)v.y;
      Bs[(n4 * 4 + 2) * 40 + kk] = (__bf16)v.z;
      Bs[(n4 * 4 + 3) * 40 + kk] = (__bf16)v.w;
    }
    waitAsync0();
    __syncthreads();
    const __bf16* arow = &As[(w * 16 + (lane & 15)) * 40];
    v16bf a = frag16(arow + kb, arow + kb + 8);
    v16bf bfr[4];
#pragma unroll
    for (int nt = 0; nt < 4; ++nt) {
      const __bf16* bcol = &Bs[(nt * 16 + (lane & 15)) * 40];
      bfr[nt] = frag16(bcol + kb, bcol + kb + 8);
    }
#pragma unroll
    for (int nt = 0; nt < 4; ++nt) acc[nt] = wmma_bf16(a, bfr[nt], acc[nt]);
  }
#pragma unroll
  for (int nt = 0; nt < 4; ++nt) {
    int n = N0 + nt * 16 + (lane & 15);
    float bias = bo[n];
#pragma unroll
    for (int j = 0; j < 8; ++j) {
      int m = M0 + w * 16 + j + ((lane >> 4) << 3);
      Out[(size_t)m * HID_ + n] = acc[nt][j] + bias;
    }
  }
}

extern "C" void kernel_launch(void* const* d_in, const int* in_sizes, int n_in,
                              void* d_out, int out_size, void* d_ws, size_t ws_size,
                              hipStream_t stream) {
  const float* x    = (const float*)d_in[0];
  const float* mask = (const float*)d_in[1];
  const float* Wq   = (const float*)d_in[2];
  const float* bq   = (const float*)d_in[3];
  const float* Wk   = (const float*)d_in[4];
  const float* bk   = (const float*)d_in[5];
  const float* Wv   = (const float*)d_in[6];
  const float* bv   = (const float*)d_in[7];
  const float* Wo   = (const float*)d_in[8];
  const float* bo   = (const float*)d_in[9];
  float* out = (float*)d_out;

  char* ws = (char*)d_ws;
  const size_t SZ = (size_t)B_ * S_ * HID_ * 2;  // 100,663,296 bytes per tensor
  __bf16* Qb  = (__bf16*)(ws);
  __bf16* Kt  = (__bf16*)(ws + SZ);
  __bf16* Vb  = (__bf16*)(ws + 2 * SZ);
  __bf16* Ctx = (__bf16*)(ws + 3 * SZ);
  __bf16* Kv  = (__bf16*)(ws + 4 * SZ);
  __bf16* Ks  = (__bf16*)(ws + 4 * SZ + (size_t)B_ * H_ * DH_ * DH_ * 2);

  k_qkv<<<dim3(HID_ / 64, (B_ * S_) / 128, 3), 256, 0, stream>>>(
      x, mask, Wq, bq, Wk, bk, Wv, bv, Qb, Kt, Vb);
  k_kv<<<dim3(B_ * H_), 512, 0, stream>>>(Kt, Vb, Kv, Ks);
  k_attn<<<dim3(S_ / 128, B_ * H_), 256, 0, stream>>>(Qb, Kv, Ks, Ctx);
  k_out<<<dim3(HID_ / 64, (B_ * S_) / 128), 256, 0, stream>>>(Ctx, Wo, bo, out);
}
